// TextEncoder_16484084483024
// MI455X (gfx1250) — matrix operand order repaired and checked by reading, not compiled, not run on hardware
//
#include <hip/hip_runtime.h>
#include <hip/hip_bf16.h>
#include <math.h>

// ---------------- problem constants (match reference) ----------------
#define BB   8
#define LL   512
#define CC   512
#define HH   8
#define FF   2048
#define NCB  4
#define NAB  6
#define KCV  5
#define WWIN 4
#define SST  50
#define HD   64          // CC/HH
#define RREL 9           // 2*WWIN+1
#define PADC 2
#define H2N  2
#define HD2N 256         // CC/H2N
#define EPSV 1e-6f

typedef __attribute__((ext_vector_type(16))) _Float16 v16h;
typedef __attribute__((ext_vector_type(8)))  _Float16 v8h;
typedef __attribute__((ext_vector_type(4)))  _Float16 v4h;
typedef __attribute__((ext_vector_type(8)))  float    v8f;

#define LDSS 40   // LDS row stride in halfs (80B = 20 banks; conflict-free b128 reads)

// =====================================================================
// LDS-tiled batched WMMA GEMM:
//   Y[m,n] = act(alpha*(sum_k A[m,k]*B(n,k) + bias[n]))*gamma[n] + Res[m,n]
//   B(n,k) = bTrans ? Bp[k*ldb + n] : Bp[n*ldb + k]
// Block tile 128(M) x 128(N), K-step 32. 8 wave32s as 2(M) x 4(N);
// each wave: 64x32 strip = 4 A-frags x 2 B-frags -> 8 v_wmma per K-step.
// Staging issues all 8 global_load_b128 first, then converts/stores, so
// loadcnt waits are staggered instead of serial full drains.
// No divergent control flow in the hot loop (EXEC all-ones for WMMA).
// =====================================================================
__global__ __launch_bounds__(256)
void k_gemm(const float* __restrict__ Ap, const float* __restrict__ Bp,
            const float* __restrict__ biasp, const float* __restrict__ gammap,
            const float* __restrict__ resp, float* __restrict__ Yp,
            int M, int N, int K, int lda, int ldb, int ldy,
            long sAb, long sAh, long sBb, long sBh, long sYb, long sYh,
            int Hdiv, float alpha, int act, int bTrans)
{
    __shared__ _Float16 ldsA[128 * LDSS];
    __shared__ _Float16 ldsB[128 * LDSS];

    int z  = blockIdx.z;
    int bb = z / Hdiv;
    int hh = z - bb * Hdiv;
    const float* A  = Ap + (long)bb * sAb + (long)hh * sAh;
    const float* Bm = Bp + (long)bb * sBb + (long)hh * sBh;
    float*       Y  = Yp + (long)bb * sYb + (long)hh * sYh;
    const float* R  = resp ? (resp + (long)bb * sYb + (long)hh * sYh) : nullptr;

    const int tid  = threadIdx.x;
    const int lane = tid & 31;
    const int wave = tid >> 5;
    const int wm   = wave >> 2;               // 0..1
    const int wn   = wave & 3;                // 0..3
    const int mBlock = blockIdx.y * 128;
    const int nBlock = blockIdx.x * 128;

    // staging index precompute (A / non-trans B)
    const int srow0 = tid >> 3;               // 0..31
    const int sk4   = (tid & 7) * 4;          // 0,4,...,28
    // staging index (trans B)
    const int tk0   = tid >> 4;               // 0..15
    const int tn0   = (tid & 15) * 4;         // 0..60

    const int lo_a = (lane < 16) ? 0 : 8;     // A-frag K offset per lane half
    const int lo_b = (lane < 16) ? 0 : 8;     // B-frag K offset per lane half
    const int lr   = lane & 15;

    // clamped global row/col bases (loop-invariant)
    int growv[4], gcolv[4];
    #pragma unroll
    for (int p = 0; p < 4; ++p) {
        int g = mBlock + srow0 + p * 32; growv[p] = (g > M - 1) ? (M - 1) : g;
        g = nBlock + srow0 + p * 32;     gcolv[p] = (g > N - 1) ? (N - 1) : g;
    }
    int gnv[2];
    #pragma unroll
    for (int pn = 0; pn < 2; ++pn) {
        int g = nBlock + tn0 + pn * 64;  gnv[pn] = (g > N - 4) ? (N - 4) : g;
    }

    v8f acc[8] = {};

    for (int kk = 0; kk < K; kk += 32) {
        // uniform, branch-free next-tile prefetch offset (always a valid address)
        const int pkk = (kk + 32 < K) ? (kk + 32) : kk;

        // ---- issue ALL staging loads first (8 loads in flight) ----
        float4 va[4];
        #pragma unroll
        for (int p = 0; p < 4; ++p) {
            va[p] = *(const float4*)(A + (long)growv[p] * lda + kk + sk4);
            __builtin_prefetch(A + (long)growv[p] * lda + pkk + sk4, 0, 1);
        }
        float4 vb[4];
        if (!bTrans) {
            #pragma unroll
            for (int p = 0; p < 4; ++p) {
                vb[p] = *(const float4*)(Bm + (long)gcolv[p] * ldb + kk + sk4);
                __builtin_prefetch(Bm + (long)gcolv[p] * ldb + pkk + sk4, 0, 1);
            }
        } else {
            #pragma unroll
            for (int pk = 0; pk < 2; ++pk) {
                #pragma unroll
                for (int pn = 0; pn < 2; ++pn) {
                    int k = tk0 + pk * 16;
                    vb[pk * 2 + pn] = *(const float4*)(Bm + (long)(kk + k) * ldb + gnv[pn]);
                    __builtin_prefetch(Bm + (long)(pkk + k) * ldb + gnv[pn], 0, 1);
                }
            }
        }

        // ---- convert + store to LDS (waits stagger as data arrives) ----
        #pragma unroll
        for (int p = 0; p < 4; ++p) {
            int row = srow0 + p * 32;
            v4h h; h[0] = (_Float16)va[p].x; h[1] = (_Float16)va[p].y;
                   h[2] = (_Float16)va[p].z; h[3] = (_Float16)va[p].w;
            *(v4h*)(ldsA + row * LDSS + sk4) = h;
        }
        if (!bTrans) {
            #pragma unroll
            for (int p = 0; p < 4; ++p) {
                int col = srow0 + p * 32;
                v4h h; h[0] = (_Float16)vb[p].x; h[1] = (_Float16)vb[p].y;
                       h[2] = (_Float16)vb[p].z; h[3] = (_Float16)vb[p].w;
                *(v4h*)(ldsB + col * LDSS + sk4) = h;
            }
        } else {
            #pragma unroll
            for (int pk = 0; pk < 2; ++pk) {
                #pragma unroll
                for (int pn = 0; pn < 2; ++pn) {
                    int k  = tk0 + pk * 16;
                    int n4 = tn0 + pn * 64;
                    const float4 v = vb[pk * 2 + pn];
                    ldsB[(n4 + 0) * LDSS + k] = (_Float16)v.x;
                    ldsB[(n4 + 1) * LDSS + k] = (_Float16)v.y;
                    ldsB[(n4 + 2) * LDSS + k] = (_Float16)v.z;
                    ldsB[(n4 + 3) * LDSS + k] = (_Float16)v.w;
                }
            }
        }
        __syncthreads();

        // ---- fragments from LDS (2x ds_load_b128 each, ISA operand layout) ----
        v16h af[4];
        #pragma unroll
        for (int im = 0; im < 4; ++im) {
            int row = wm * 64 + im * 16 + lr;
            v8h a0 = *(const v8h*)(ldsA + row * LDSS + lo_a);
            v8h a1 = *(const v8h*)(ldsA + row * LDSS + lo_a + 16);
            af[im] = __builtin_shufflevector(a0, a1, 0,1,2,3,4,5,6,7,8,9,10,11,12,13,14,15);
        }
        v16h bf[2];
        #pragma unroll
        for (int jn = 0; jn < 2; ++jn) {
            int col = wn * 32 + jn * 16 + lr;
            v8h b0 = *(const v8h*)(ldsB + col * LDSS + lo_b);
            v8h b1 = *(const v8h*)(ldsB + col * LDSS + lo_b + 16);
            bf[jn] = __builtin_shufflevector(b0, b1, 0,1,2,3,4,5,6,7,8,9,10,11,12,13,14,15);
        }

        #pragma unroll
        for (int im = 0; im < 4; ++im)
            #pragma unroll
            for (int jn = 0; jn < 2; ++jn)
                acc[im * 2 + jn] = __builtin_amdgcn_wmma_f32_16x16x32_f16(
                    false, af[im], false, bf[jn], (short)0, acc[im * 2 + jn],
                    false, false);
        __syncthreads();
    }

    // ---- epilogue ----
    #pragma unroll
    for (int im = 0; im < 4; ++im) {
        #pragma unroll
        for (int jn = 0; jn < 2; ++jn) {
            int n = nBlock + wn * 32 + jn * 16 + lr;
            if (n < N) {
                float bias = biasp  ? biasp[n]  : 0.0f;
                float gam  = gammap ? gammap[n] : 1.0f;
                #pragma unroll
                for (int r = 0; r < 8; ++r) {
                    int m = mBlock + wm * 64 + im * 16 + r + ((lane < 16) ? 0 : 8);
                    if (m < M) {
                        float y = (acc[im * 2 + jn][r] + bias) * alpha;
                        if      (act == 1) y = 0.5f * y * (1.0f + erff(y * 0.70710678118654752f));
                        else if (act == 2) y = fmaxf(y, 0.0f);
                        else if (act == 3) y = tanhf(y);
                        y *= gam;
                        if (R) y += R[(long)m * ldy + n];
                        Y[(long)m * ldy + n] = y;
                    }
                }
            }
        }
    }
}

// ---------------- elementwise / small kernels ----------------

__global__ void k_embed(const int* __restrict__ tok, const float* __restrict__ emb,
                        float* __restrict__ x)
{
    int i = blockIdx.x * blockDim.x + threadIdx.x;
    if (i >= BB * LL * CC) return;
    int c  = i % CC;
    int bl = i / CC;
    x[i] = emb[(long)tok[bl] * CC + c];
}

__global__ void k_mask(float* __restrict__ x, const float* __restrict__ mask)
{
    int i = blockIdx.x * blockDim.x + threadIdx.x;
    if (i >= BB * LL * CC) return;
    int bl = i / CC;
    x[i] *= mask[bl];
}

__global__ void k_dwconv(const float* __restrict__ x, const float* __restrict__ w,
                         const float* __restrict__ bias, const float* __restrict__ mask,
                         float* __restrict__ y)
{
    int i = blockIdx.x * blockDim.x + threadIdx.x;
    if (i >= BB * LL * CC) return;
    int c = i % CC;
    int l = (i / CC) % LL;
    int b = i / (CC * LL);
    float s = 0.0f;
    #pragma unroll
    for (int k = 0; k < KCV; ++k) {
        int lp = l + k - PADC;
        lp = lp < 0 ? 0 : (lp >= LL ? LL - 1 : lp);
        s += x[((long)b * LL + lp) * CC + c] * mask[b * LL + lp] * w[c * KCV + k];
    }
    y[i] = (s + bias[c]) * mask[b * LL + l];
}

__global__ __launch_bounds__(256)
void k_layernorm(const float* __restrict__ x, const float* __restrict__ g,
                 const float* __restrict__ b, float* __restrict__ y)
{
    int row = blockIdx.x;                  // over B*L rows
    const float* xr = x + (long)row * CC;
    __shared__ float red[256];
    int tid = threadIdx.x;

    float s = 0.0f;
    for (int c = tid; c < CC; c += 256) s += xr[c];
    red[tid] = s; __syncthreads();
    for (int o = 128; o > 0; o >>= 1) { if (tid < o) red[tid] += red[tid + o]; __syncthreads(); }
    float mean = red[0] / (float)CC;
    __syncthreads();

    float v = 0.0f;
    for (int c = tid; c < CC; c += 256) { float d = xr[c] - mean; v += d * d; }
    red[tid] = v; __syncthreads();
    for (int o = 128; o > 0; o >>= 1) { if (tid < o) red[tid] += red[tid + o]; __syncthreads(); }
    float rstd = rsqrtf(red[0] / (float)CC + EPSV);
    __syncthreads();

    for (int c = tid; c < CC; c += 256)
        y[(long)row * CC + c] = (xr[c] - mean) * rstd * g[c] + b[c];
}

// scores[b,h,l,m] += dot(q[b,l,h,:], rel_k[r,:]) for m = l + r - W
__global__ void k_relk(const float* __restrict__ q, const float* __restrict__ relk,
                       float* __restrict__ scores)
{
    int i = blockIdx.x * blockDim.x + threadIdx.x;
    if (i >= BB * HH * LL * RREL) return;
    int r = i % RREL;
    int l = (i / RREL) % LL;
    int h = (i / (RREL * LL)) % HH;
    int b = i / (RREL * LL * HH);
    int m = l + r - WWIN;
    if (m < 0 || m >= LL) return;
    const float* qp = q + ((long)b * LL + l) * CC + h * HD;
    const float* rp = relk + r * HD;
    float s = 0.0f;
    #pragma unroll 8
    for (int d = 0; d < HD; ++d) s += qp[d] * rp[d];
    scores[(((long)b * HH + h) * LL + l) * LL + m] += s;
}

__global__ __launch_bounds__(256)
void k_softmax(float* __restrict__ scores, const float* __restrict__ mask)
{
    long row = blockIdx.x;                 // (b*H + h)*L + l
    int  l   = (int)(row % LL);
    int  b   = (int)(row / ((long)HH * LL));
    float* sr = scores + row * LL;
    float qm = mask[b * LL + l];
    __shared__ float red[256];
    int tid = threadIdx.x;

    float mx = -1e30f;
    for (int m = tid; m < LL; m += 256) {
        float v = sr[m];
        if (qm == 0.0f || mask[b * LL + m] == 0.0f) v = -10000.0f;
        sr[m] = v;
        mx = fmaxf(mx, v);
    }
    red[tid] = mx; __syncthreads();
    for (int o = 128; o > 0; o >>= 1) { if (tid < o) red[tid] = fmaxf(red[tid], red[tid + o]); __syncthreads(); }
    mx = red[0]; __syncthreads();

    float sum = 0.0f;
    for (int m = tid; m < LL; m += 256) { float e = expf(sr[m] - mx); sr[m] = e; sum += e; }
    red[tid] = sum; __syncthreads();
    for (int o = 128; o > 0; o >>= 1) { if (tid < o) red[tid] += red[tid + o]; __syncthreads(); }
    float inv = 1.0f / red[0];
    __syncthreads();
    for (int m = tid; m < LL; m += 256) sr[m] *= inv;
}

// out[b,l,h*HD+d] += sum_r attn[b,h,l,l+r-W] * rel_v[r,d]
__global__ void k_relv(const float* __restrict__ attn, const float* __restrict__ relv,
                       float* __restrict__ o)
{
    int i = blockIdx.x * blockDim.x + threadIdx.x;
    if (i >= BB * HH * LL * HD) return;
    int d = i % HD;
    int l = (i / HD) % LL;
    int h = (i / (HD * LL)) % HH;
    int b = i / (HD * LL * HH);
    const float* ar = attn + (((long)b * HH + h) * LL + l) * LL;
    float s = 0.0f;
    #pragma unroll
    for (int r = 0; r < RREL; ++r) {
        int m = l + r - WWIN;
        if (m >= 0 && m < LL) s += ar[m] * relv[r * HD + d];
    }
    o[((long)b * LL + l) * CC + h * HD + d] += s;
}

// ---------------- style attention helpers ----------------

__global__ void k_stylesc(const float* __restrict__ q, const float* __restrict__ kk,
                          float* __restrict__ sc)
{
    int i = blockIdx.x * blockDim.x + threadIdx.x;
    if (i >= BB * LL * H2N * SST) return;
    int s  = i % SST;
    int h2 = (i / SST) % H2N;
    int bt = i / (SST * H2N);
    const float* qp = q + (long)bt * CC + h2 * HD2N;
    const float* kp = kk + (long)s * CC + h2 * HD2N;
    float acc = 0.0f;
    #pragma unroll 8
    for (int d = 0; d < HD2N; ++d) acc += qp[d] * kp[d];
    sc[i] = acc * 0.044194173824159216f;   // C^-0.5
}

__global__ void k_stylesm(float* __restrict__ sc, const float* __restrict__ mask)
{
    int i = blockIdx.x * blockDim.x + threadIdx.x;
    if (i >= BB * LL * H2N) return;
    int bt = i / H2N;
    float* r = sc + (long)i * SST;
    float mx = -1e30f;
    for (int s = 0; s < SST; ++s) mx = fmaxf(mx, r[s]);
    float sum = 0.0f;
    for (int s = 0; s < SST; ++s) { float e = expf(r[s] - mx); r[s] = e; sum += e; }
    float mk = (mask[bt] == 0.0f) ? 0.0f : 1.0f;
    float inv = mk / sum;
    for (int s = 0; s < SST; ++s) r[s] *= inv;
}

__global__ void k_styleo(const float* __restrict__ sc, const float* __restrict__ vv,
                         float* __restrict__ o)
{
    int i = blockIdx.x * blockDim.x + threadIdx.x;
    if (i >= BB * LL * CC) return;
    int c  = i % CC;
    int bt = i / CC;
    int h2 = c / HD2N;
    const float* a = sc + ((long)bt * H2N + h2) * SST;
    float acc = 0.0f;
    #pragma unroll 10
    for (int s = 0; s < SST; ++s) acc += a[s] * vv[(long)s * CC + c];
    o[i] = acc;
}

__global__ void k_transpose_out(const float* __restrict__ x, const float* __restrict__ mask,
                                float* __restrict__ y)
{
    int i = blockIdx.x * blockDim.x + threadIdx.x;      // over B*C*L
    if (i >= BB * CC * LL) return;
    int l = i % LL;
    int c = (i / LL) % CC;
    int b = i / (LL * CC);
    y[i] = x[((long)b * LL + l) * CC + c] * mask[b * LL + l];
}

// =====================================================================
// host orchestration
// =====================================================================
static void gemm(hipStream_t st, const float* A, const float* Bm, const float* bias,
                 const float* gamma, const float* res, float* Y,
                 int M, int N, int K, int lda, int ldb, int ldy,
                 long sAb, long sAh, long sBb, long sBh, long sYb, long sYh,
                 int Hdiv, int Z, float alpha, int act, int bT)
{
    dim3 g((N + 127) / 128, (M + 127) / 128, Z);
    k_gemm<<<g, 256, 0, st>>>(A, Bm, bias, gamma, res, Y, M, N, K, lda, ldb, ldy,
                              sAb, sAh, sBb, sBh, sYb, sYh, Hdiv, alpha, act, bT);
}

extern "C" void kernel_launch(void* const* d_in, const int* in_sizes, int n_in,
                              void* d_out, int out_size, void* d_ws, size_t ws_size,
                              hipStream_t stream)
{
    (void)in_sizes; (void)n_in; (void)out_size; (void)ws_size;
    // ---- inputs (setup_inputs order) ----
    const int*   tokens   = (const int*)  d_in[0];
    const float* style_v  = (const float*)d_in[1];
    const float* mask     = (const float*)d_in[2];
    const float* embed    = (const float*)d_in[3];
    const float* dw_w     = (const float*)d_in[4];
    const float* dw_b     = (const float*)d_in[5];
    const float* cln_g    = (const float*)d_in[6];
    const float* cln_b    = (const float*)d_in[7];
    const float* pw1_w    = (const float*)d_in[8];
    const float* pw1_b    = (const float*)d_in[9];
    const float* pw2_w    = (const float*)d_in[10];
    const float* pw2_b    = (const float*)d_in[11];
    const float* cgamma   = (const float*)d_in[12];
    const float* aq_w     = (const float*)d_in[13];
    const float* aq_b     = (const float*)d_in[14];
    const float* ak_w     = (const float*)d_in[15];
    const float* ak_b     = (const float*)d_in[16];
    const float* av_w     = (const float*)d_in[17];
    const float* av_b     = (const float*)d_in[18];
    const float* ao_w     = (const float*)d_in[19];
    const float* ao_b     = (const float*)d_in[20];
    const float* relk     = (const float*)d_in[21];
    const float* relv     = (const float*)d_in[22];
    const float* ln1_g    = (const float*)d_in[23];
    const float* ln1_b    = (const float*)d_in[24];
    const float* ln2_g    = (const float*)d_in[25];
    const float* ln2_b    = (const float*)d_in[26];
    const float* f1_w     = (const float*)d_in[27];
    const float* f1_b     = (const float*)d_in[28];
    const float* f2_w     = (const float*)d_in[29];
    const float* f2_b     = (const float*)d_in[30];
    const float* skey     = (const float*)d_in[31];
    const float* sq_w     = (const float*)d_in[32];
    const float* sq_b     = (const float*)d_in[33];
    const float* sk_w     = (const float*)d_in[34];
    const float* sk_b     = (const float*)d_in[35];
    const float* sv_w     = (const float*)d_in[36];
    const float* sv_b     = (const float*)d_in[37];
    const float* so_w     = (const float*)d_in[38];
    const float* so_b     = (const float*)d_in[39];
    const float* sn_g     = (const float*)d_in[40];
    const float* sn_b     = (const float*)d_in[41];
    float* out = (float*)d_out;

    // ---- workspace carve-up ----
    const long BLC = (long)BB * LL * CC;           // 2,097,152
    float* W = (float*)d_ws;
    size_t off = 0;
    float* xbuf   = W + off; off += BLC;
    float* tmp    = W + off; off += BLC;
    float* qbuf   = W + off; off += BLC;
    float* kbuf   = W + off; off += BLC;
    float* vbuf   = W + off; off += BLC;
    float* obuf   = W + off; off += BLC;
    float* hbuf   = W + off; off += (long)BB * LL * FF;
    float* scores = W + off; off += (long)BB * HH * LL * LL;
    float* kkb    = W + off; off += (long)SST * CC;
    float* vvb    = W + off; off += (long)SST * CC;
    float* scb    = W + off; off += (long)BB * LL * H2N * SST;
    float* sxbuf  = hbuf;    // x1 for style layer 2 (hbuf free at that point)

    const int ML = BB * LL;                        // 4096
    const int EW = (BB * LL * CC + 255) / 256;     // elementwise grid

    // ---- embedding ----
    k_embed<<<EW, 256, 0, stream>>>(tokens, embed, xbuf);

    // ---- ConvNeXt stack ----
    for (int i = 0; i < NCB; ++i) {
        k_mask<<<EW, 256, 0, stream>>>(xbuf, mask);
        k_dwconv<<<EW, 256, 0, stream>>>(xbuf, dw_w + (long)i * CC * KCV,
                                         dw_b + (long)i * CC, mask, tmp);
        k_layernorm<<<ML, 256, 0, stream>>>(tmp, cln_g + (long)i * CC,
                                            cln_b + (long)i * CC, tmp);
        gemm(stream, tmp, pw1_w + (long)i * FF * CC, pw1_b + (long)i * FF,
             nullptr, nullptr, hbuf, ML, FF, CC, CC, CC, FF,
             0, 0, 0, 0, 0, 0, 1, 1, 1.0f, /*gelu*/1, 0);
        gemm(stream, hbuf, pw2_w + (long)i * CC * FF, pw2_b + (long)i * CC,
             cgamma + (long)i * CC, xbuf, xbuf, ML, CC, FF, FF, FF, CC,
             0, 0, 0, 0, 0, 0, 1, 1, 1.0f, 0, 0);
        k_mask<<<EW, 256, 0, stream>>>(xbuf, mask);
    }

    // ---- windowed relative-position attention stack ----
    const float qscale = 0.125f;                   // HD^-0.5
    for (int i = 0; i < NAB; ++i) {
        gemm(stream, xbuf, aq_w + (long)i * CC * CC, aq_b + (long)i * CC,
             nullptr, nullptr, qbuf, ML, CC, CC, CC, CC, CC,
             0, 0, 0, 0, 0, 0, 1, 1, qscale, 0, 0);
        gemm(stream, xbuf, ak_w + (long)i * CC * CC, ak_b + (long)i * CC,
             nullptr, nullptr, kbuf, ML, CC, CC, CC, CC, CC,
             0, 0, 0, 0, 0, 0, 1, 1, 1.0f, 0, 0);
        gemm(stream, xbuf, av_w + (long)i * CC * CC, av_b + (long)i * CC,
             nullptr, nullptr, vbuf, ML, CC, CC, CC, CC, CC,
             0, 0, 0, 0, 0, 0, 1, 1, 1.0f, 0, 0);
        // scores = q @ k^T, batched over (b,h)
        gemm(stream, qbuf, kbuf, nullptr, nullptr, nullptr, scores,
             LL, LL, HD, CC, CC, LL,
             (long)LL * CC, HD, (long)LL * CC, HD,
             (long)HH * LL * LL, (long)LL * LL, HH, BB * HH, 1.0f, 0, 0);
        {
            int n = BB * HH * LL * RREL;
            k_relk<<<(n + 255) / 256, 256, 0, stream>>>(qbuf, relk + (long)i * RREL * HD, scores);
        }
        k_softmax<<<BB * HH * LL, 256, 0, stream>>>(scores, mask);
        // out = attn @ v, batched over (b,h); v accessed transposed
        gemm(stream, scores, vbuf, nullptr, nullptr, nullptr, obuf,
             LL, HD, LL, LL, CC, CC,
             (long)HH * LL * LL, (long)LL * LL, (long)LL * CC, HD,
             (long)LL * CC, HD, HH, BB * HH, 1.0f, 0, /*bTrans*/1);
        {
            int n = BB * HH * LL * HD;
            k_relv<<<(n + 255) / 256, 256, 0, stream>>>(scores, relv + (long)i * RREL * HD, obuf);
        }
        gemm(stream, obuf, ao_w + (long)i * CC * CC, ao_b + (long)i * CC,
             nullptr, xbuf, tmp, ML, CC, CC, CC, CC, CC,
             0, 0, 0, 0, 0, 0, 1, 1, 1.0f, 0, 0);
        k_layernorm<<<ML, 256, 0, stream>>>(tmp, ln1_g + (long)i * CC, ln1_b + (long)i * CC, xbuf);
        gemm(stream, xbuf, f1_w + (long)i * FF * CC, f1_b + (long)i * FF,
             nullptr, nullptr, hbuf, ML, FF, CC, CC, CC, FF,
             0, 0, 0, 0, 0, 0, 1, 1, 1.0f, /*relu*/2, 0);
        gemm(stream, hbuf, f2_w + (long)i * CC * FF, f2_b + (long)i * CC,
             nullptr, xbuf, tmp, ML, CC, FF, FF, FF, CC,
             0, 0, 0, 0, 0, 0, 1, 1, 1.0f, 0, 0);
        k_layernorm<<<ML, 256, 0, stream>>>(tmp, ln2_g + (long)i * CC, ln2_b + (long)i * CC, xbuf);
    }
    k_mask<<<EW, 256, 0, stream>>>(xbuf, mask);

    // ---- style attention (2 layers, residual from original xt each time) ----
    for (int si = 0; si < 2; ++si) {
        const float* xin  = (si == 0) ? xbuf : sxbuf;
        float*       xout = (si == 0) ? sxbuf : tmp;
        gemm(stream, skey, sk_w + (long)si * CC * CC, sk_b + (long)si * CC,
             nullptr, nullptr, kkb, SST, CC, CC, CC, CC, CC,
             0, 0, 0, 0, 0, 0, 1, 1, 1.0f, /*tanh*/3, 0);
        gemm(stream, style_v, sv_w + (long)si * CC * CC, sv_b + (long)si * CC,
             nullptr, nullptr, vvb, SST, CC, CC, CC, CC, CC,
             0, 0, 0, 0, 0, 0, 1, 1, 1.0f, 0, 0);
        gemm(stream, xin, sq_w + (long)si * CC * CC, sq_b + (long)si * CC,
             nullptr, nullptr, qbuf, ML, CC, CC, CC, CC, CC,
             0, 0, 0, 0, 0, 0, 1, 1, 1.0f, 0, 0);
        {
            int n = BB * LL * H2N * SST;
            k_stylesc<<<(n + 255) / 256, 256, 0, stream>>>(qbuf, kkb, scb);
        }
        {
            int n = BB * LL * H2N;
            k_stylesm<<<(n + 255) / 256, 256, 0, stream>>>(scb, mask);
        }
        k_styleo<<<EW, 256, 0, stream>>>(scb, vvb, obuf);
        gemm(stream, obuf, so_w + (long)si * CC * CC, so_b + (long)si * CC,
             nullptr, xbuf /*residual = original xt*/, xout, ML, CC, CC, CC, CC, CC,
             0, 0, 0, 0, 0, 0, 1, 1, 1.0f, 0, 0);
    }
    k_layernorm<<<ML, 256, 0, stream>>>(tmp, sn_g, sn_b, tmp);
    {
        int n = BB * CC * LL;
        k_transpose_out<<<(n + 255) / 256, 256, 0, stream>>>(tmp, mask, out);
    }
}
